// DualChannelTransformer_50981261804086
// MI455X (gfx1250) — matrix operand order repaired and checked by reading, not compiled, not run on hardware
//
#include <hip/hip_runtime.h>
#include <hip/hip_bf16.h>

// ---------------- problem constants ----------------
#define BB 4
#define SS 1024
#define DD 512
#define HH 8
#define LL 4
#define FF 2048
#define DK 64
#define MM (BB * SS)   // 4096 rows for all big GEMMs

typedef __bf16 bf16;
typedef bf16  v16bf __attribute__((ext_vector_type(16)));
typedef float v8f   __attribute__((ext_vector_type(8)));

// ---- WMMA fragment loaders (contiguous k order shared by A and B, wave32) ----
// A-matrix 16x32 bf16: lane L<16 -> row M=L, K=k0..k0+15 contiguous, k0=0
//                      lane L>=16 -> row M=L-16, k0=16
__device__ __forceinline__ v16bf ldfragA(const bf16* p) {
  union { v16bf v; uint4 q[2]; } u;
  u.q[0] = *(const uint4*)(p);        // 8 bf16
  u.q[1] = *(const uint4*)(p + 8);    // 8 bf16
  return u.v;
}
// B-matrix 32x16 bf16: lane L<16 -> col N=L, K=0..15 contiguous; lane>=16 -> K=16..31
__device__ __forceinline__ v16bf ldfragB(const bf16* p) {
  union { v16bf v; uint4 q[2]; } u;
  u.q[0] = *(const uint4*)(p);
  u.q[1] = *(const uint4*)(p + 8);
  return u.v;
}
__device__ __forceinline__ void st8(bf16* dst, float4 a, float4 b) {
  union { bf16 h[8]; uint4 q; } u;
  u.h[0] = (bf16)a.x; u.h[1] = (bf16)a.y; u.h[2] = (bf16)a.z; u.h[3] = (bf16)a.w;
  u.h[4] = (bf16)b.x; u.h[5] = (bf16)b.y; u.h[6] = (bf16)b.z; u.h[7] = (bf16)b.w;
  *(uint4*)dst = u.q;
}
__device__ __forceinline__ v8f wmma_bf16(v16bf a, v16bf b, v8f c) {
  return __builtin_amdgcn_wmma_f32_16x16x32_bf16(false, a, false, b,
                                                 (short)0, c, false, false);
}

// =================== bf16 WMMA GEMM ===================
// C[M,N] = A[M,K] @ W[K,N] + bias  (A,W fp32 in HBM, converted to bf16 in LDS)
// Tile 128x64x32, double-buffered LDS; 8 waves in 4x2 grid, each wave owns a
// 32x32 C block (2x2 accumulators -> 4 WMMAs per K-step, 1 barrier per step).
__global__ __launch_bounds__(256) void k_gemm(const float* __restrict__ A,
    const float* __restrict__ W, const float* __restrict__ bias,
    float* __restrict__ C, int N, int K, int relu)
{
  __shared__ bf16 sA[2][128][40];   // +8 pad keeps 16B align, spreads banks
  __shared__ bf16 sB[2][64][40];    // stored transposed: sB[buf][n][k]

  const int tid  = threadIdx.x;
  const int lane = tid & 31, w = tid >> 5;
  const int wm = w & 3, wn = w >> 2;          // 4x2 wave grid
  const int lr = lane & 15;
  const int k0 = (lane & 16) ? 16 : 0;        // A-fragment half select
  const int kb16 = lane & 16;                 // B-fragment base (0 or 16)
  const int rowBase = blockIdx.x * 128;
  const int colBase = blockIdx.y * 64;

  const int ar = tid >> 1, ac = (tid & 1) * 16;  // A stage: 128 rows x 32 k
  const int bk = tid >> 3, bn = (tid & 7) * 8;   // W stage: 32 k x 64 n

  v8f acc[2][2] = {};

  auto stage = [&](int kb, int buf) {
    const float* ga = A + (size_t)(rowBase + ar) * K + kb + ac;
    st8(&sA[buf][ar][ac],     *(const float4*)ga,       *(const float4*)(ga + 4));
    st8(&sA[buf][ar][ac + 8], *(const float4*)(ga + 8), *(const float4*)(ga + 12));
    const float* gb = W + (size_t)(kb + bk) * N + colBase + bn;
    float4 x = *(const float4*)gb, y = *(const float4*)(gb + 4);
    sB[buf][bn + 0][bk] = (bf16)x.x; sB[buf][bn + 1][bk] = (bf16)x.y;
    sB[buf][bn + 2][bk] = (bf16)x.z; sB[buf][bn + 3][bk] = (bf16)x.w;
    sB[buf][bn + 4][bk] = (bf16)y.x; sB[buf][bn + 5][bk] = (bf16)y.y;
    sB[buf][bn + 6][bk] = (bf16)y.z; sB[buf][bn + 7][bk] = (bf16)y.w;
    if (kb + 32 < K) {   // keep next tiles flowing into the WGP caches
      __builtin_prefetch(A + (size_t)(rowBase + ar) * K + kb + 32 + ac, 0, 3);
      __builtin_prefetch(W + (size_t)(kb + 32 + bk) * N + colBase + bn, 0, 3);
    }
  };

  stage(0, 0);
  int cur = 0;
  for (int kb = 0; kb < K; kb += 32) {
    __syncthreads();                       // buf[cur] ready; buf[cur^1] free
    if (kb + 32 < K) stage(kb + 32, cur ^ 1);
    v16bf a0 = ldfragA(&sA[cur][wm * 32 + lr][k0]);
    v16bf a1 = ldfragA(&sA[cur][wm * 32 + 16 + lr][k0]);
    v16bf b0 = ldfragB(&sB[cur][wn * 32 + lr][kb16]);
    v16bf b1 = ldfragB(&sB[cur][wn * 32 + 16 + lr][kb16]);
    acc[0][0] = wmma_bf16(a0, b0, acc[0][0]);
    acc[0][1] = wmma_bf16(a0, b1, acc[0][1]);
    acc[1][0] = wmma_bf16(a1, b0, acc[1][0]);
    acc[1][1] = wmma_bf16(a1, b1, acc[1][1]);
    cur ^= 1;
  }
  const int m0 = (lane & 16) >> 1;            // 0 or 8
#pragma unroll
  for (int ni = 0; ni < 2; ++ni) {
    const int cn = colBase + wn * 32 + ni * 16 + lr;
    const float bv = bias ? bias[cn] : 0.f;
#pragma unroll
    for (int mi = 0; mi < 2; ++mi) {
#pragma unroll
      for (int j = 0; j < 8; ++j) {
        float v = acc[mi][ni][j] + bv;
        if (relu) v = fmaxf(v, 0.f);
        C[(size_t)(rowBase + wm * 32 + mi * 16 + m0 + j) * N + cn] = v;
      }
    }
  }
}

// =================== attention ===================
// One workgroup per (b, h, 32-row q tile). Full-row softmax over S=1024.
// K-chunk and V-chunk staging double-buffered against the WMMAs.
__global__ __launch_bounds__(256) void k_attn(const float* __restrict__ Q,
    const float* __restrict__ Km, const float* __restrict__ V,
    float* __restrict__ O)
{
  extern __shared__ char smem[];
  float* sS = (float*)smem;                 // [32][1024] f32 scores  (128 KB)
  bf16*  sP = (bf16*)(sS + 32 * 1024);      // [32][1024] bf16 probs ( 64 KB)
  bf16*  sQ = sP + 32 * 1024;               // [32][72]
  bf16*  sK = sQ + 32 * 72;                 // [2][64][72]  (reused as V^T)

  const int tid = threadIdx.x;
  const int lane = tid & 31, w = tid >> 5;
  const int lr = lane & 15;
  const int k0 = (lane & 16) ? 16 : 0;
  const int kb16 = lane & 16;

  int idx = blockIdx.x;
  const int qc = idx & 31; idx >>= 5;       // S/32 = 32 q tiles
  const int h  = idx & 7;  const int b = idx >> 3;
  const int qBase = qc * 32;
  const size_t headOff = ((size_t)b * SS) * DD + h * DK;

  // stage Q tile 32x64 fp32 -> bf16
  {
    const int r = tid >> 3, c = (tid & 7) * 8;
    const float* gp = Q + headOff + (size_t)(qBase + r) * DD + c;
    st8(&sQ[r * 72 + c], *(const float4*)gp, *(const float4*)(gp + 4));
  }

  const int qs = w & 1, ks = w >> 1;        // wave -> 16x16 score tile
  const int m0 = (lane & 16) >> 1;

  // ---- scores: S = (Q K^T) * 1/sqrt(dk) ----
  auto stageK = [&](int kc, int buf) {
    const int r = tid >> 2, c = (tid & 3) * 16;
    const float* gp = Km + headOff + (size_t)(kc * 64 + r) * DD + c;
    bf16* dst = sK + (size_t)buf * 64 * 72 + r * 72 + c;
    st8(dst,     *(const float4*)gp,       *(const float4*)(gp + 4));
    st8(dst + 8, *(const float4*)(gp + 8), *(const float4*)(gp + 12));
  };
  stageK(0, 0);
  int cur = 0;
  for (int kc = 0; kc < SS / 64; ++kc) {
    __syncthreads();
    if (kc + 1 < SS / 64) stageK(kc + 1, cur ^ 1);
    const bf16* kbuf = sK + (size_t)cur * 64 * 72;
    v8f acc = {};
#pragma unroll
    for (int s = 0; s < 2; ++s) {   // d_k = 64 = 2 x 32
      v16bf aq = ldfragA(&sQ[(qs * 16 + lr) * 72 + s * 32 + k0]);
      v16bf bk = ldfragB(&kbuf[(ks * 16 + lr) * 72 + s * 32 + kb16]);
      acc = wmma_bf16(aq, bk, acc);
    }
#pragma unroll
    for (int j = 0; j < 8; ++j)
      sS[(qs * 16 + m0 + j) * 1024 + kc * 64 + ks * 16 + lr] = acc[j] * 0.125f;
    cur ^= 1;
  }
  __syncthreads();

  // ---- softmax: 4 rows per wave, shfl reductions (wave32) ----
  for (int rr = 0; rr < 4; ++rr) {
    const int r = w * 4 + rr;
    float* row = sS + (size_t)r * 1024;
    float mx = -3.4e38f;
    for (int k = lane; k < 1024; k += 32) mx = fmaxf(mx, row[k]);
#pragma unroll
    for (int m = 16; m >= 1; m >>= 1) mx = fmaxf(mx, __shfl_xor(mx, m, 32));
    float sum = 0.f;
    for (int k = lane; k < 1024; k += 32) {
      float e = __expf(row[k] - mx); row[k] = e; sum += e;
    }
#pragma unroll
    for (int m = 16; m >= 1; m >>= 1) sum += __shfl_xor(sum, m, 32);
    const float inv = 1.f / sum;
    bf16* prow = sP + (size_t)r * 1024;
    for (int k = lane; k < 1024; k += 32) prow[k] = (bf16)(row[k] * inv);
  }
  __syncthreads();

  // ---- ctx = P @ V : wave -> 16 q rows x 16 d cols ----
  const int ds = w >> 1;
  auto stageV = [&](int kc, int buf) {   // transpose: sVt[d][kk]
    const int kk = tid >> 2, d0 = (tid & 3) * 16;
    const float* gp = V + headOff + (size_t)(kc * 64 + kk) * DD + d0;
    bf16* dst = sK + (size_t)buf * 64 * 72;
#pragma unroll
    for (int jj = 0; jj < 4; ++jj) {
      float4 t4 = *(const float4*)(gp + jj * 4);
      dst[(d0 + jj * 4 + 0) * 72 + kk] = (bf16)t4.x;
      dst[(d0 + jj * 4 + 1) * 72 + kk] = (bf16)t4.y;
      dst[(d0 + jj * 4 + 2) * 72 + kk] = (bf16)t4.z;
      dst[(d0 + jj * 4 + 3) * 72 + kk] = (bf16)t4.w;
    }
  };
  v8f oacc = {};
  stageV(0, 0);
  cur = 0;
  for (int kc = 0; kc < SS / 64; ++kc) {
    __syncthreads();
    if (kc + 1 < SS / 64) stageV(kc + 1, cur ^ 1);
    const bf16* vbuf = sK + (size_t)cur * 64 * 72;
#pragma unroll
    for (int s = 0; s < 2; ++s) {
      v16bf ap = ldfragA(&sP[(qs * 16 + lr) * 1024 + kc * 64 + s * 32 + k0]);
      v16bf bv = ldfragB(&vbuf[(ds * 16 + lr) * 72 + s * 32 + kb16]);
      oacc = wmma_bf16(ap, bv, oacc);
    }
    cur ^= 1;
  }
#pragma unroll
  for (int j = 0; j < 8; ++j)
    O[headOff + (size_t)(qBase + qs * 16 + m0 + j) * DD + ds * 16 + lr] = oacc[j];
}

// =================== residual + LayerNorm ===================
__global__ __launch_bounds__(256) void k_add_ln(const float* __restrict__ y,
    const float* __restrict__ res, const float* __restrict__ g,
    const float* __restrict__ bt, float* __restrict__ o)
{
  __shared__ float red[8];
  const int row = blockIdx.x, t = threadIdx.x;
  const int lane = t & 31, w = t >> 5;
  const size_t base = (size_t)row * DD;
  float v0 = y[base + t] + res[base + t];
  float v1 = y[base + t + 256] + res[base + t + 256];
  float s = v0 + v1;
#pragma unroll
  for (int m = 16; m >= 1; m >>= 1) s += __shfl_xor(s, m, 32);
  if (lane == 0) red[w] = s;
  __syncthreads();
  float tot = 0.f;
#pragma unroll
  for (int i = 0; i < 8; ++i) tot += red[i];
  const float mean = tot * (1.f / DD);
  __syncthreads();
  float d0 = v0 - mean, d1 = v1 - mean;
  float q = d0 * d0 + d1 * d1;
#pragma unroll
  for (int m = 16; m >= 1; m >>= 1) q += __shfl_xor(q, m, 32);
  if (lane == 0) red[w] = q;
  __syncthreads();
  float vq = 0.f;
#pragma unroll
  for (int i = 0; i < 8; ++i) vq += red[i];
  const float rstd = rsqrtf(vq * (1.f / DD) + 1e-5f);
  o[base + t]       = d0 * rstd * g[t] + bt[t];
  o[base + t + 256] = d1 * rstd * g[t + 256] + bt[t + 256];
}

// =================== embed: X[B,S,6] @ W[6,D] + b + pe ===================
__global__ __launch_bounds__(256) void k_embed(const float* __restrict__ X,
    const float* __restrict__ W, const float* __restrict__ bias,
    const float* __restrict__ pe, float* __restrict__ out)
{
  const int idx = blockIdx.x * 256 + threadIdx.x;   // MM*DD threads
  const int bs = idx / DD, d = idx % DD;
  const int s = bs & (SS - 1);
  float acc = bias[d] + pe[(size_t)s * DD + d];
#pragma unroll
  for (int i = 0; i < 6; ++i) acc += X[(size_t)bs * 6 + i] * W[i * DD + d];
  out[idx] = acc;
}

// =================== mean pool over seq -> fused[B, 2D] ===================
__global__ __launch_bounds__(256) void k_pool(const float* __restrict__ l,
    const float* __restrict__ r, float* __restrict__ fused)
{
  const int idx = blockIdx.x * 256 + threadIdx.x;
  if (idx >= BB * 2 * DD) return;
  const int b = idx / (2 * DD), d2 = idx % (2 * DD);
  const float* src = (d2 < DD) ? l : r;
  const int d = (d2 < DD) ? d2 : d2 - DD;
  float acc = 0.f;
  for (int s = 0; s < SS; ++s) acc += src[((size_t)b * SS + s) * DD + d];
  fused[idx] = acc * (1.f / SS);
}

// =================== head MLPs (tiny) ===================
__global__ __launch_bounds__(256) void k_heads(const float* __restrict__ fused,
    const float* w1a, const float* b1a, const float* w2a, const float* b2a,
    const float* w1b, const float* b1b, const float* w2b, const float* b2b,
    float* __restrict__ out)
{
  __shared__ float hid[BB][DD];
  const int t = threadIdx.x;
  for (int hh = 0; hh < 2; ++hh) {
    const float* w1 = hh ? w1b : w1a; const float* b1 = hh ? b1b : b1a;
    const float* w2 = hh ? w2b : w2a; const float* b2 = hh ? b2b : b2a;
    for (int i = t; i < BB * DD; i += 256) {
      const int b = i / DD, d = i % DD;
      float acc = b1[d];
      for (int k = 0; k < 2 * DD; ++k)
        acc += fused[b * 2 * DD + k] * w1[(size_t)k * DD + d];
      hid[b][d] = fmaxf(acc, 0.f);
    }
    __syncthreads();
    if (t < BB * 2) {
      const int b = t / 2, oo = t % 2;
      float acc = b2[oo];
      for (int k = 0; k < DD; ++k) acc += hid[b][k] * w2[k * 2 + oo];
      out[hh * (BB * 2) + t] = acc;
    }
    __syncthreads();
  }
}

// =================== orchestration ===================
extern "C" void kernel_launch(void* const* d_in, const int* in_sizes, int n_in,
                              void* d_out, int out_size, void* d_ws, size_t ws_size,
                              hipStream_t stream) {
  (void)in_sizes; (void)n_in; (void)out_size; (void)ws_size;
  const float* left_w  = (const float*)d_in[0];
  const float* right_w = (const float*)d_in[1];
  const float* Wl = (const float*)d_in[2];  const float* bl = (const float*)d_in[3];
  const float* Wr = (const float*)d_in[4];  const float* br = (const float*)d_in[5];
  const float* pe = (const float*)d_in[6];
  const float* mha_w = (const float*)d_in[7];
  const float* mha_b = (const float*)d_in[8];
  const float* mha_g = (const float*)d_in[9];
  const float* mha_bt = (const float*)d_in[10];
  const float* ff_w1 = (const float*)d_in[11]; const float* ff_b1 = (const float*)d_in[12];
  const float* ff_w2 = (const float*)d_in[13]; const float* ff_b2 = (const float*)d_in[14];
  const float* ff_g = (const float*)d_in[15];  const float* ff_bt = (const float*)d_in[16];
  const float* h1w1 = (const float*)d_in[17]; const float* h1b1 = (const float*)d_in[18];
  const float* h1w2 = (const float*)d_in[19]; const float* h1b2 = (const float*)d_in[20];
  const float* h2w1 = (const float*)d_in[21]; const float* h2b1 = (const float*)d_in[22];
  const float* h2w2 = (const float*)d_in[23]; const float* h2b2 = (const float*)d_in[24];
  float* out = (float*)d_out;

  float* ws = (float*)d_ws;
  const size_t SZ = (size_t)MM * DD;
  float* bufAl = ws;            float* bufAr = ws + SZ;
  float* bufBl = ws + 2 * SZ;   float* bufBr = ws + 3 * SZ;
  float* lc    = ws + 4 * SZ;   float* rc    = ws + 5 * SZ;
  float* tq    = ws + 6 * SZ;   float* tk    = ws + 7 * SZ;
  float* tv    = ws + 8 * SZ;   float* tctx  = ws + 9 * SZ;
  float* tgo   = ws + 10 * SZ;
  float* hbuf  = ws + 11 * SZ;                 // [MM, FF]
  float* fused = hbuf + (size_t)MM * FF;       // [BB, 2*DD]

  auto gemm = [&](const float* A, const float* W, const float* bias, float* C,
                  int N, int K, int relu) {
    k_gemm<<<dim3(MM / 128, N / 64), 256, 0, stream>>>(A, W, bias, C, N, K, relu);
  };
  const size_t attnSmem = 32 * 1024 * 4 + 32 * 1024 * 2 + 32 * 72 * 2
                        + 2 * 64 * 72 * 2;
  auto attn = [&](const float* Q, const float* K, const float* V, float* O) {
    k_attn<<<dim3(BB * HH * (SS / 32)), 256, attnSmem, stream>>>(Q, K, V, O);
  };
  auto addln = [&](const float* y, const float* res, const float* g,
                   const float* bt, float* o) {
    k_add_ln<<<dim3(MM), 256, 0, stream>>>(y, res, g, bt, o);
  };
  auto mha = [&](const float* xq, const float* xkv, int l, int blk, float* ob) {
    const float* w  = mha_w + ((size_t)(l * 4 + blk)) * 4 * DD * DD;
    const float* bs = mha_b + ((size_t)(l * 4 + blk)) * 4 * DD;
    gemm(xq,  w + 0 * (size_t)DD * DD, bs + 0 * DD, tq, DD, DD, 0);
    gemm(xkv, w + 1 * (size_t)DD * DD, bs + 1 * DD, tk, DD, DD, 0);
    gemm(xkv, w + 2 * (size_t)DD * DD, bs + 2 * DD, tv, DD, DD, 0);
    attn(tq, tk, tv, tctx);
    gemm(tctx, w + 3 * (size_t)DD * DD, bs + 3 * DD, tgo, DD, DD, 0);
    addln(tgo, xq, mha_g + (size_t)(l * 4 + blk) * DD,
          mha_bt + (size_t)(l * 4 + blk) * DD, ob);
  };
  auto ffn = [&](const float* x, int l, int c, float* ob) {
    gemm(x, ff_w1 + (size_t)(l * 2 + c) * DD * FF, ff_b1 + (size_t)(l * 2 + c) * FF,
         hbuf, FF, DD, 1);
    gemm(hbuf, ff_w2 + (size_t)(l * 2 + c) * FF * DD, ff_b2 + (size_t)(l * 2 + c) * DD,
         tgo, DD, FF, 0);
    addln(tgo, x, ff_g + (size_t)(l * 2 + c) * DD, ff_bt + (size_t)(l * 2 + c) * DD, ob);
  };

  // embed both channels
  k_embed<<<dim3(MM * DD / 256), 256, 0, stream>>>(left_w, Wl, bl, pe, bufAl);
  k_embed<<<dim3(MM * DD / 256), 256, 0, stream>>>(right_w, Wr, br, pe, bufAr);

  float* cl = bufAl; float* cr = bufAr; float* nl = bufBl; float* nr = bufBr;
  for (int l = 0; l < LL; ++l) {
    mha(cl, cr, l, 0, lc);   // cross 1<-2
    mha(cr, cl, l, 1, rc);   // cross 2<-1
    mha(lc, lc, l, 2, cl);   // self 1 (overwrite cl, no longer needed)
    mha(rc, rc, l, 3, cr);   // self 2
    ffn(cl, l, 0, nl);
    ffn(cr, l, 1, nr);
    float* t;
    t = cl; cl = nl; nl = t;
    t = cr; cr = nr; nr = t;
  }

  k_pool<<<dim3((BB * 2 * DD + 255) / 256), 256, 0, stream>>>(cl, cr, fused);
  k_heads<<<dim3(1), 256, 0, stream>>>(fused, h1w1, h1b1, h1w2, h1b2,
                                       h2w1, h2b1, h2w2, h2b2, out);
}
